// GSATransformerBlock_42949672960317
// MI455X (gfx1250) — matrix operand order repaired and checked by reading, not compiled, not run on hardware
//
#include <hip/hip_runtime.h>

// ---------------------------------------------------------------------------
// GSA transformer block for MI455X (gfx1250, wave32, WMMA 16x16x32 bf16).
// Banded attention (|i-j|<=48) -> 128-wide key window per 16-query tile.
// MLP GEMMs stage B-tiles through LDS with the Tensor Data Mover
// (tensor_load_to_lds + s_wait_tensorcnt), double buffered.
// ---------------------------------------------------------------------------

#define NTOK 4096
#define CIN 512
#define COUT 512
#define HIDD 128
#define NHEAD 4
#define MIDD 2048
#define SEQL 48
#define NEGV -1e9f

typedef __attribute__((ext_vector_type(16))) __bf16 v16bf;
typedef __attribute__((ext_vector_type(8)))  __bf16 v8bf;
typedef __attribute__((ext_vector_type(8)))  float  v8f;
typedef __attribute__((ext_vector_type(4)))  unsigned v4u;
typedef __attribute__((ext_vector_type(8)))  int      v8i;
typedef __attribute__((ext_vector_type(4)))  int      v4i;

#if defined(__HIP_DEVICE_COMPILE__) && __has_builtin(__builtin_amdgcn_tensor_load_to_lds) && __has_builtin(__builtin_amdgcn_s_wait_tensorcnt)
#define HAS_TDM 1
#else
#define HAS_TDM 0
#endif

__device__ __forceinline__ __bf16 f2bf(float f) {
  unsigned u = __builtin_bit_cast(unsigned, f);
  unsigned r = u + 0x7FFFu + ((u >> 16) & 1u);   // round-to-nearest-even
  unsigned short h = (unsigned short)(r >> 16);
  return __builtin_bit_cast(__bf16, h);
}

__device__ __forceinline__ v8f vzero8() {
  v8f z;
#pragma unroll
  for (int i = 0; i < 8; ++i) z[i] = 0.f;
  return z;
}

// A-fragment (16x32 bf16, row-major source, leading dim lda).
__device__ __forceinline__ v16bf load_a_frag(const __bf16* base, int lda) {
  int lane = threadIdx.x & 31;
  return *(const v16bf*)(base + (lane & 15) * lda + ((lane >> 4) << 4));
}

// B-fragment (32x16 bf16) from column-major storage Bc[col][k].
__device__ __forceinline__ v16bf load_b_frag(const __bf16* colbase, int ldb) {
  int lane = threadIdx.x & 31;
  return *(const v16bf*)(colbase + (lane & 15) * ldb + ((lane >> 4) << 4));
}

__device__ __forceinline__ v8f wmma_bf16(v16bf a, v16bf b, v8f c) {
  return __builtin_amdgcn_wmma_f32_16x16x32_bf16(
      false, a, false, b, (short)0, c, false, false);
}

#if HAS_TDM
// Issue a 2D TDM load: tile_w x tile_h (elements, bf16) from a row-major
// tensor of width tensor_w / height tensor_h, row stride = tensor_w, into
// LDS at byte offset lds_off.  D# layout per CDNA5 ISA ch.8.
__device__ __forceinline__ void tdm_load_tile(unsigned lds_off, const void* gaddr,
                                              unsigned tensor_w, unsigned tensor_h,
                                              unsigned tile_w, unsigned tile_h) {
  unsigned long long ga = (unsigned long long)gaddr;
  v4u g0;
  g0[0] = 1u;                                         // count=1, no gather
  g0[1] = lds_off;                                    // lds_addr
  g0[2] = (unsigned)(ga & 0xFFFFFFFFu);               // global_addr[31:0]
  g0[3] = (unsigned)((ga >> 32) & 0x01FFFFFFu) | (2u << 30);  // addr[56:32], type=2
  v8i g1;
  unsigned long long s0 = tensor_w;                   // dim0 stride (elements)
  g1[0] = (int)0x00010000u;                           // data_size=1 (2 bytes)
  g1[1] = (int)((tensor_w & 0xFFFFu) << 16);          // tensor_dim0 lo16 @bit48
  g1[2] = (int)((tensor_w >> 16) | ((tensor_h & 0xFFFFu) << 16));
  g1[3] = (int)((tensor_h >> 16) | (tile_w << 16));   // tile_dim0 @bit112
  g1[4] = (int)(tile_h & 0xFFFFu);                    // tile_dim1; tile_dim2=0
  g1[5] = (int)(s0 & 0xFFFFFFFFu);                    // tensor_dim0_stride lo32
  g1[6] = (int)((s0 >> 32) & 0xFFFFu);                // stride hi16; dim1_stride=0
  g1[7] = 0;
  v4i z4 = {0, 0, 0, 0};
#if __clang_major__ >= 23
  v8i z8 = {0, 0, 0, 0, 0, 0, 0, 0};
  __builtin_amdgcn_tensor_load_to_lds(g0, g1, z4, z4, z8, 0);
#else
  __builtin_amdgcn_tensor_load_to_lds(g0, g1, z4, z4, 0);
#endif
}
#endif

// ---------------------------------------------------------------------------
// Weight transpose + fp32 -> bf16 cast: in[K][Nc] (row-major) -> out[Nc][K]
// ---------------------------------------------------------------------------
__global__ void transpose_cast_kernel(const float* __restrict__ in,
                                      __bf16* __restrict__ out,
                                      int K, int Nc) {
  const float* ib = in + (long long)blockIdx.z * K * Nc;
  __bf16* ob = out + (long long)blockIdx.z * K * Nc;
  int idx = blockIdx.x * blockDim.x + threadIdx.x;
  if (idx < K * Nc) {
    int k = idx / Nc, n = idx % Nc;
    ob[(long long)n * K + k] = f2bf(ib[idx]);
  }
}

// ---------------------------------------------------------------------------
// LayerNorm (+ optional ReLU) -> bf16. One block (256 threads) per row.
// ---------------------------------------------------------------------------
template <bool RELU>
__global__ __launch_bounds__(256) void ln_cast_kernel(
    const float* __restrict__ in, const float* __restrict__ g,
    const float* __restrict__ b, __bf16* __restrict__ out, int C) {
  int row = blockIdx.x;
  const float* x = in + (long long)row * C;
  int tid = threadIdx.x;
  float s = 0.f, s2 = 0.f;
  for (int c = tid; c < C; c += 256) { float v = x[c]; s += v; s2 += v * v; }
#pragma unroll
  for (int m = 1; m < 32; m <<= 1) {
    s += __shfl_xor(s, m, 32);
    s2 += __shfl_xor(s2, m, 32);
  }
  __shared__ float ws1[8], ws2[8];
  if ((tid & 31) == 0) { ws1[tid >> 5] = s; ws2[tid >> 5] = s2; }
  __syncthreads();
  float ts = 0.f, ts2 = 0.f;
#pragma unroll
  for (int w = 0; w < 8; ++w) { ts += ws1[w]; ts2 += ws2[w]; }
  float mean = ts / (float)C;
  float var = ts2 / (float)C - mean * mean;
  float inv = rsqrtf(var + 1e-5f);
  for (int c = tid; c < C; c += 256) {
    float v = (x[c] - mean) * inv * g[c] + b[c];
    if (RELU) v = fmaxf(v, 0.f);
    out[(long long)row * C + c] = f2bf(v);
  }
}

// ---------------------------------------------------------------------------
// Batched WMMA GEMM (B from global): C[M][Nc] = A[M][K] x Bc[col][k].
// 4 waves/block, each wave a 16x64 strip.  Used for QKV projections.
// ---------------------------------------------------------------------------
template <bool BF16_OUT, bool TRANS_OUT>
__global__ __launch_bounds__(128) void gemm_wmma_kernel(
    const __bf16* __restrict__ A, const __bf16* __restrict__ B, void* Cout,
    int M, int Nc, int Kd, long long sB, long long sC) {
  int wave = threadIdx.x >> 5;
  int lane = threadIdx.x & 31;
  int row0 = blockIdx.x * 64 + wave * 16;
  int col0 = blockIdx.y * 64;
  const __bf16* Bb = B + (long long)blockIdx.z * sB;
  long long cz = (long long)blockIdx.z * sC;

  v8f acc[4];
#pragma unroll
  for (int t = 0; t < 4; ++t) acc[t] = vzero8();

  for (int k0 = 0; k0 < Kd; k0 += 32) {
    v16bf a = load_a_frag(A + (long long)row0 * Kd + k0, Kd);
    v16bf b0 = load_b_frag(Bb + (long long)(col0 +  0) * Kd + k0, Kd);
    v16bf b1 = load_b_frag(Bb + (long long)(col0 + 16) * Kd + k0, Kd);
    v16bf b2 = load_b_frag(Bb + (long long)(col0 + 32) * Kd + k0, Kd);
    v16bf b3 = load_b_frag(Bb + (long long)(col0 + 48) * Kd + k0, Kd);
    acc[0] = wmma_bf16(a, b0, acc[0]);
    acc[1] = wmma_bf16(a, b1, acc[1]);
    acc[2] = wmma_bf16(a, b2, acc[2]);
    acc[3] = wmma_bf16(a, b3, acc[3]);
  }

  int cN = lane & 15;
  int rb = (lane >> 4) << 3;
#pragma unroll
  for (int t = 0; t < 4; ++t) {
#pragma unroll
    for (int r = 0; r < 8; ++r) {
      int row = row0 + rb + r;
      int col = col0 + t * 16 + cN;
      float v = acc[t][r];
      if (TRANS_OUT) {
        ((__bf16*)Cout)[cz + (long long)col * M + row] = f2bf(v);
      } else if (BF16_OUT) {
        ((__bf16*)Cout)[cz + (long long)row * Nc + col] = f2bf(v);
      } else {
        ((float*)Cout)[cz + (long long)row * Nc + col] = v;
      }
    }
  }
}

// ---------------------------------------------------------------------------
// WMMA GEMM with TDM-staged B tiles (double-buffered 64x32 bf16 in LDS).
// Wave 0 drives the Tensor Data Mover; workgroup barriers hand off buffers.
// f32 output with optional residual.  Used for the MLP GEMMs.
// ---------------------------------------------------------------------------
template <bool RESIDUAL>
__global__ __launch_bounds__(128) void gemm_wmma_tdm_kernel(
    const __bf16* __restrict__ A, const __bf16* __restrict__ Bc,
    float* __restrict__ Cout, const float* __restrict__ Res,
    int M, int Nc, int Kd) {
  __shared__ alignas(128) __bf16 btile[2][64 * 32];
  int wave = threadIdx.x >> 5;
  int lane = threadIdx.x & 31;
  int row0 = blockIdx.x * 64 + wave * 16;
  int col0 = blockIdx.y * 64;
  int S = Kd >> 5;

  v8f acc[4];
#pragma unroll
  for (int t = 0; t < 4; ++t) acc[t] = vzero8();

#if HAS_TDM
  if (wave == 0) {
    tdm_load_tile((unsigned)(unsigned long long)(void*)&btile[0][0],
                  Bc + (long long)col0 * Kd, (unsigned)Kd, (unsigned)Nc, 32u, 64u);
  }
#endif

  for (int s = 0; s < S; ++s) {
    int buf = s & 1;
#if HAS_TDM
    if (wave == 0) {
      if (s + 1 < S) {
        tdm_load_tile((unsigned)(unsigned long long)(void*)&btile[buf ^ 1][0],
                      Bc + (long long)col0 * Kd + (s + 1) * 32,
                      (unsigned)Kd, (unsigned)Nc, 32u, 64u);
        __builtin_amdgcn_s_wait_tensorcnt(1);   // current tile landed
      } else {
        __builtin_amdgcn_s_wait_tensorcnt(0);
      }
    }
    __syncthreads();
#else
    // fallback: cooperative copy of the 64x32 tile
    __syncthreads();
    {
      int r = threadIdx.x >> 1, half = threadIdx.x & 1;
      const __bf16* src = Bc + (long long)(col0 + r) * Kd + s * 32 + half * 16;
      *(v16bf*)&btile[buf][r * 32 + half * 16] = *(const v16bf*)src;
    }
    __syncthreads();
#endif

    v16bf a = load_a_frag(A + (long long)row0 * Kd + s * 32, Kd);
    v16bf b0 = load_b_frag(&btile[buf][ 0 * 32], 32);
    v16bf b1 = load_b_frag(&btile[buf][16 * 32], 32);
    v16bf b2 = load_b_frag(&btile[buf][32 * 32], 32);
    v16bf b3 = load_b_frag(&btile[buf][48 * 32], 32);
    acc[0] = wmma_bf16(a, b0, acc[0]);
    acc[1] = wmma_bf16(a, b1, acc[1]);
    acc[2] = wmma_bf16(a, b2, acc[2]);
    acc[3] = wmma_bf16(a, b3, acc[3]);
    __syncthreads();   // done with btile[buf] before it is refilled
  }

  int cN = lane & 15;
  int rb = (lane >> 4) << 3;
#pragma unroll
  for (int t = 0; t < 4; ++t) {
#pragma unroll
    for (int r = 0; r < 8; ++r) {
      int row = row0 + rb + r;
      int col = col0 + t * 16 + cN;
      float v = acc[t][r];
      if (RESIDUAL) v += Res[(long long)row * Nc + col];
      Cout[(long long)row * Nc + col] = v;
    }
  }
}

// ---------------------------------------------------------------------------
// Fused banded attention: one wave per (head, 16-query tile).  96 WMMAs/wave.
// ---------------------------------------------------------------------------
__global__ __launch_bounds__(128) void attn_kernel(
    const __bf16* __restrict__ Q, const __bf16* __restrict__ Km,
    const __bf16* __restrict__ Vt, const __bf16* __restrict__ WoT,
    const float* __restrict__ X, const float* __restrict__ pos,
    const float* __restrict__ ori, const int* __restrict__ seq,
    const int* __restrict__ batch, float* __restrict__ attn_x) {
  __shared__ alignas(32) __bf16 smem[4][2][16 * 128];
  int wave = threadIdx.x >> 5;
  int lane = threadIdx.x & 31;
  int h = blockIdx.y;
  int i0 = (blockIdx.x * 4 + wave) * 16;

  const __bf16* Qh = Q + (long long)h * NTOK * HIDD;
  const __bf16* Kh = Km + (long long)h * NTOK * HIDD;
  const __bf16* Vh = Vt + (long long)h * HIDD * NTOK;  // [d][token]
  const __bf16* Wh = WoT + h * HIDD * HIDD;            // [e][d]

  int j0 = i0 - SEQL;
  if (j0 < 0) j0 = 0;
  if (j0 > NTOK - 128) j0 = NTOK - 128;

  // ---- scores = Q * K^T / sqrt(HID) -----------------------------------
  v16bf qa[4];
#pragma unroll
  for (int kk = 0; kk < 4; ++kk)
    qa[kk] = load_a_frag(Qh + (long long)i0 * HIDD + kk * 32, HIDD);

  v8f sc[8];
#pragma unroll
  for (int kt = 0; kt < 8; ++kt) {
    const __bf16* kb = Kh + (long long)(j0 + kt * 16) * HIDD;
    v16bf b0 = load_b_frag(kb + 0, HIDD);
    v16bf b1 = load_b_frag(kb + 32, HIDD);
    v16bf b2 = load_b_frag(kb + 64, HIDD);
    v16bf b3 = load_b_frag(kb + 96, HIDD);
    v8f c = vzero8();
    c = wmma_bf16(qa[0], b0, c);
    c = wmma_bf16(qa[1], b1, c);
    c = wmma_bf16(qa[2], b2, c);
    c = wmma_bf16(qa[3], b3, c);
    sc[kt] = c;
  }

  int cN = lane & 15;
  int hi = lane >> 4;

  float pix[8], piy[8], piz[8], oix[8], oiy[8], oiz[8];
  int bi[8], si[8];
#pragma unroll
  for (int r = 0; r < 8; ++r) {
    int i = i0 + hi * 8 + r;
    pix[r] = pos[i * 3 + 0]; piy[r] = pos[i * 3 + 1]; piz[r] = pos[i * 3 + 2];
    oix[r] = ori[i * 3 + 0]; oiy[r] = ori[i * 3 + 1]; oiz[r] = ori[i * 3 + 2];
    bi[r] = batch[i]; si[r] = seq[i];
  }

  const float scale = rsqrtf((float)HIDD);
  float mrow[8], srow[8];
#pragma unroll
  for (int r = 0; r < 8; ++r) mrow[r] = -1e30f;

#pragma unroll
  for (int kt = 0; kt < 8; ++kt) {
    int j = j0 + kt * 16 + cN;
    float pjx = pos[j * 3 + 0], pjy = pos[j * 3 + 1], pjz = pos[j * 3 + 2];
    float ojx = ori[j * 3 + 0], ojy = ori[j * 3 + 1], ojz = ori[j * 3 + 2];
    int bj = batch[j], sj = seq[j];
#pragma unroll
    for (int r = 0; r < 8; ++r) {
      float s = sc[kt][r] * scale + (oix[r] * ojx + oiy[r] * ojy + oiz[r] * ojz);
      float dx = pix[r] - pjx, dy = piy[r] - pjy, dz = piz[r] - pjz;
      float d2 = dx * dx + dy * dy + dz * dz;
      int ds = si[r] - sj; ds = ds < 0 ? -ds : ds;
      bool ok = (bi[r] == bj) && (d2 <= 1.0f) && (ds <= SEQL);
      s = ok ? s : NEGV;
      sc[kt][r] = s;
      mrow[r] = fmaxf(mrow[r], s);
    }
  }
#pragma unroll
  for (int r = 0; r < 8; ++r) {
#pragma unroll
    for (int m = 1; m < 16; m <<= 1) mrow[r] = fmaxf(mrow[r], __shfl_xor(mrow[r], m, 16));
  }
#pragma unroll
  for (int r = 0; r < 8; ++r) srow[r] = 0.f;
#pragma unroll
  for (int kt = 0; kt < 8; ++kt) {
#pragma unroll
    for (int r = 0; r < 8; ++r) {
      float p = __expf(sc[kt][r] - mrow[r]);
      sc[kt][r] = p;
      srow[r] += p;
    }
  }
#pragma unroll
  for (int r = 0; r < 8; ++r) {
#pragma unroll
    for (int m = 1; m < 16; m <<= 1) srow[r] += __shfl_xor(srow[r], m, 16);
    srow[r] = 1.f / srow[r];
  }

  __bf16* P = &smem[wave][0][0];
#pragma unroll
  for (int kt = 0; kt < 8; ++kt)
#pragma unroll
    for (int r = 0; r < 8; ++r)
      P[(hi * 8 + r) * 128 + kt * 16 + cN] = f2bf(sc[kt][r] * srow[r]);
  __syncthreads();

  // ---- O = P * V ------------------------------------------------------
  __bf16* O = &smem[wave][1][0];
  v16bf pa[4];
#pragma unroll
  for (int kk = 0; kk < 4; ++kk) pa[kk] = load_a_frag(P + kk * 32, 128);
#pragma unroll
  for (int dt = 0; dt < 8; ++dt) {
    const __bf16* vb = Vh + (long long)(dt * 16) * NTOK + j0;
    v16bf b0 = load_b_frag(vb + 0, NTOK);
    v16bf b1 = load_b_frag(vb + 32, NTOK);
    v16bf b2 = load_b_frag(vb + 64, NTOK);
    v16bf b3 = load_b_frag(vb + 96, NTOK);
    v8f c = vzero8();
    c = wmma_bf16(pa[0], b0, c);
    c = wmma_bf16(pa[1], b1, c);
    c = wmma_bf16(pa[2], b2, c);
    c = wmma_bf16(pa[3], b3, c);
#pragma unroll
    for (int r = 0; r < 8; ++r)
      O[(hi * 8 + r) * 128 + dt * 16 + cN] = f2bf(c[r]);
  }
  __syncthreads();

  // ---- out = O * Wo + x (residual) ------------------------------------
  v16bf oa[4];
#pragma unroll
  for (int kk = 0; kk < 4; ++kk) oa[kk] = load_a_frag(O + kk * 32, 128);
#pragma unroll
  for (int et = 0; et < 8; ++et) {
    const __bf16* wb = Wh + (long long)(et * 16) * HIDD;
    v16bf b0 = load_b_frag(wb + 0, HIDD);
    v16bf b1 = load_b_frag(wb + 32, HIDD);
    v16bf b2 = load_b_frag(wb + 64, HIDD);
    v16bf b3 = load_b_frag(wb + 96, HIDD);
    v8f c = vzero8();
    c = wmma_bf16(oa[0], b0, c);
    c = wmma_bf16(oa[1], b1, c);
    c = wmma_bf16(oa[2], b2, c);
    c = wmma_bf16(oa[3], b3, c);
#pragma unroll
    for (int r = 0; r < 8; ++r) {
      int i = i0 + hi * 8 + r;
      int col = h * HIDD + et * 16 + cN;
      attn_x[(long long)i * COUT + col] = c[r] + X[(long long)i * COUT + col];
    }
  }
}

// ---------------------------------------------------------------------------
extern "C" void kernel_launch(void* const* d_in, const int* in_sizes, int n_in,
                              void* d_out, int out_size, void* d_ws, size_t ws_size,
                              hipStream_t stream) {
  (void)in_sizes; (void)n_in; (void)out_size; (void)ws_size;
  const float* x     = (const float*)d_in[0];
  const float* pos   = (const float*)d_in[1];
  const float* ori   = (const float*)d_in[2];
  const int*   seq   = (const int*)d_in[3];
  const int*   batch = (const int*)d_in[4];
  const float* ln1_g = (const float*)d_in[5];
  const float* ln1_b = (const float*)d_in[6];
  const float* ln2_g = (const float*)d_in[7];
  const float* ln2_b = (const float*)d_in[8];
  const float* Wq    = (const float*)d_in[9];
  const float* Wk    = (const float*)d_in[10];
  const float* Wv    = (const float*)d_in[11];
  const float* Wo    = (const float*)d_in[12];
  const float* lnm_g = (const float*)d_in[13];
  const float* lnm_b = (const float*)d_in[14];
  const float* W1    = (const float*)d_in[15];
  const float* W2    = (const float*)d_in[16];
  float* out = (float*)d_out;

  char* ws = (char*)d_ws;
  size_t off = 0;
  auto alloc = [&](size_t bytes) -> void* {
    off = (off + 255) & ~(size_t)255;
    void* p = ws + off;
    off += bytes;
    return p;
  };

  __bf16* WqT  = (__bf16*)alloc((size_t)NHEAD * HIDD * CIN * 2);
  __bf16* WkT  = (__bf16*)alloc((size_t)NHEAD * HIDD * CIN * 2);
  __bf16* WvT  = (__bf16*)alloc((size_t)NHEAD * HIDD * CIN * 2);
  __bf16* WoT  = (__bf16*)alloc((size_t)NHEAD * HIDD * HIDD * 2);
  __bf16* W1T  = (__bf16*)alloc((size_t)MIDD * COUT * 2);
  __bf16* W2T  = (__bf16*)alloc((size_t)COUT * MIDD * 2);
  __bf16* xn   = (__bf16*)alloc((size_t)NTOK * CIN * 2);
  __bf16* Qb   = (__bf16*)alloc((size_t)NHEAD * NTOK * HIDD * 2);
  __bf16* Kb   = (__bf16*)alloc((size_t)NHEAD * NTOK * HIDD * 2);
  __bf16* Vtb  = (__bf16*)alloc((size_t)NHEAD * HIDD * NTOK * 2);
  float*  attn_x = (float*)alloc((size_t)NTOK * COUT * 4);
  __bf16* h2in = (__bf16*)alloc((size_t)NTOK * COUT * 2);
  float*  h1   = (float*)alloc((size_t)NTOK * MIDD * 4);
  __bf16* h1n  = (__bf16*)alloc((size_t)NTOK * MIDD * 2);

  // 1) weight transposes + bf16 casts
  {
    int e = CIN * HIDD;
    dim3 g((e + 255) / 256, 1, NHEAD);
    transpose_cast_kernel<<<g, 256, 0, stream>>>(Wq, WqT, CIN, HIDD);
    transpose_cast_kernel<<<g, 256, 0, stream>>>(Wk, WkT, CIN, HIDD);
    transpose_cast_kernel<<<g, 256, 0, stream>>>(Wv, WvT, CIN, HIDD);
  }
  {
    int e = HIDD * HIDD;
    dim3 g((e + 255) / 256, 1, NHEAD);
    transpose_cast_kernel<<<g, 256, 0, stream>>>(Wo, WoT, HIDD, HIDD);
  }
  transpose_cast_kernel<<<dim3((COUT * MIDD + 255) / 256, 1, 1), 256, 0, stream>>>(W1, W1T, COUT, MIDD);
  transpose_cast_kernel<<<dim3((MIDD * COUT + 255) / 256, 1, 1), 256, 0, stream>>>(W2, W2T, MIDD, COUT);

  // 2) LN1 -> xn (bf16)
  ln_cast_kernel<false><<<NTOK, 256, 0, stream>>>(x, ln1_g, ln1_b, xn, CIN);

  // 3) QKV projections (batched over heads). V stored transposed [h][d][n].
  {
    dim3 g(NTOK / 64, HIDD / 64, NHEAD);
    long long sB = (long long)HIDD * CIN;
    long long sC = (long long)NTOK * HIDD;
    gemm_wmma_kernel<true, false><<<g, 128, 0, stream>>>(
        xn, WqT, Qb, NTOK, HIDD, CIN, sB, sC);
    gemm_wmma_kernel<true, false><<<g, 128, 0, stream>>>(
        xn, WkT, Kb, NTOK, HIDD, CIN, sB, sC);
    gemm_wmma_kernel<true, true><<<g, 128, 0, stream>>>(
        xn, WvT, Vtb, NTOK, HIDD, CIN, sB, sC);
  }

  // 4) fused banded attention + Wo + residual -> attn_x (f32)
  attn_kernel<<<dim3(NTOK / 64, NHEAD, 1), 128, 0, stream>>>(
      Qb, Kb, Vtb, WoT, x, pos, ori, seq, batch, attn_x);

  // 5) LN2 -> h2in (bf16)
  ln_cast_kernel<false><<<NTOK, 256, 0, stream>>>(attn_x, ln2_g, ln2_b, h2in, COUT);

  // 6) h1 = h2in @ W1 (f32 out) — TDM-staged B tiles
  gemm_wmma_tdm_kernel<false><<<dim3(NTOK / 64, MIDD / 64, 1), 128, 0, stream>>>(
      h2in, W1T, h1, nullptr, NTOK, MIDD, COUT);

  // 7) LN(mid) + ReLU -> h1n (bf16)
  ln_cast_kernel<true><<<NTOK, 256, 0, stream>>>(h1, lnm_g, lnm_b, h1n, MIDD);

  // 8) out = h1n @ W2 + attn_x (f32) — TDM-staged B tiles
  gemm_wmma_tdm_kernel<true><<<dim3(NTOK / 64, COUT / 64, 1), 128, 0, stream>>>(
      h1n, W2T, out, attn_x, NTOK, COUT, MIDD);
}
